// TransformerLayer_25924422599005
// MI455X (gfx1250) — matrix operand order repaired and checked by reading, not compiled, not run on hardware
//
#include <hip/hip_runtime.h>
#include <math.h>

// Problem sizes (compile-time constants from the reference)
#define BQ   4
#define SQ   1024
#define DQ   1024
#define HQ   16
#define DKQ  64
#define DFFQ 4096

typedef __bf16 bf16_t;
typedef bf16_t v16bf __attribute__((ext_vector_type(16)));
typedef float  v8f   __attribute__((ext_vector_type(8)));

union FragU { v16bf v; bf16_t h[16]; uint4 q[2]; };
union AccU  { v8f   v; float  f[8]; };

__device__ __forceinline__ v8f wmma_bf16(v16bf a, v16bf b, v8f c) {
  return __builtin_amdgcn_wmma_f32_16x16x32_bf16(false, a, false, b, (short)0, c,
                                                 false, false);
}

// ---- gfx1250 async global->LDS DMA (ASYNCcnt-tracked) ----------------------
// GV addressing: VDST = LDS byte address, VADDR = 64-bit global address.
// Flat pointers to LDS carry the LDS byte offset in their low 32 bits.
__device__ __forceinline__ void async_copy_b128(const float* g, const float* l) {
  asm volatile("global_load_async_to_lds_b128 %0, %1, off"
               :
               : "v"((unsigned)(uintptr_t)l), "v"(g)
               : "memory");
}
__device__ __forceinline__ void wait_async_le8() {
  asm volatile("s_wait_asynccnt 0x8" ::: "memory");
}
__device__ __forceinline__ void wait_async_le0() {
  asm volatile("s_wait_asynccnt 0x0" ::: "memory");
}

// ---- fragment loaders (f32 LDS tile -> bf16 fragment) ----------------------
// A-matrix 16x32: lanes 0-15 rows M=0..15 hold K=0..15,
// lanes 16-31 K=16..31 (contiguous 16). Tile stored [row][32] f32.
__device__ __forceinline__ v16bf ldsFragA_f32(const float* t, int rbase) {
  const int lane = threadIdx.x & 31;
  const float* p = t + (rbase + (lane & 15)) * 32 + ((lane & 16) ? 16 : 0);
  FragU f;
#pragma unroll
  for (int e = 0; e < 8; ++e) {
    f.h[e]     = (bf16_t)p[e];
    f.h[e + 8] = (bf16_t)p[e + 8];
  }
  return f.v;
}
// B-matrix 32x16: lane holds column n=lane&15; lanes 0-15 K=0..15,
// lanes 16-31 K=16..31 (contiguous 16).
__device__ __forceinline__ v16bf ldsFragB_f32(const float* t, int rbase) {
  const int lane = threadIdx.x & 31;
  const float* p = t + (rbase + (lane & 15)) * 32 + ((lane & 16) ? 16 : 0);
  FragU f;
#pragma unroll
  for (int e = 0; e < 16; ++e) f.h[e] = (bf16_t)p[e];
  return f.v;
}
// f32 global -> bf16 A fragment (row pointer already offset by kb)
__device__ __forceinline__ v16bf globalFragA(const float* p, float s) {
  FragU f;
#pragma unroll
  for (int e = 0; e < 8; ++e) {
    f.h[e]     = (bf16_t)(p[e] * s);
    f.h[e + 8] = (bf16_t)(p[e + 8] * s);
  }
  return f.v;
}
// f32 global -> bf16 B fragment, contiguous 16
__device__ __forceinline__ v16bf globalFragB(const float* p) {
  FragU f;
#pragma unroll
  for (int e = 0; e < 16; ++e) f.h[e] = (bf16_t)p[e];
  return f.v;
}

// ---- GEMM: C[M,N] = A[M,K] @ W[N,K]^T + bias, optional ReLU ----------------
// Block tile 128x128x32, 8 waves (4x2), each wave 32x64 = 2x4 WMMA tiles.
// Double-buffered f32 LDS tiles filled by async global->LDS DMA; WMMA on the
// current stage overlaps the DMA of the next stage.
template <int RELU>
__global__ __launch_bounds__(256) void gemm_bias_kernel(
    const float* __restrict__ A, const float* __restrict__ W,
    const float* __restrict__ bias, float* __restrict__ C, int M, int N, int K) {
  constexpr int BM = 128, BN = 128, BK = 32;
  __shared__ float As[2][BM * BK];  // 16 KB per stage
  __shared__ float Bs[2][BN * BK];  // 16 KB per stage

  const int tid  = threadIdx.x;
  const int lane = tid & 31;
  const int wid  = tid >> 5;
  const int wm   = (wid >> 1) * 32;  // wave M offset in tile
  const int wn   = (wid & 1) * 64;   // wave N offset in tile
  const int m0   = blockIdx.y * BM;
  const int n0   = blockIdx.x * BN;

  AccU acc[2][4];
#pragma unroll
  for (int i = 0; i < 2; ++i)
#pragma unroll
    for (int j = 0; j < 4; ++j)
#pragma unroll
      for (int r = 0; r < 8; ++r) acc[i][j].f[r] = 0.0f;

  // Per-stage DMA: each thread issues 4 b128 copies for A and 4 for B.
  auto issue_stage = [&](int k0, int s) {
#pragma unroll
    for (int c = 0; c < 4; ++c) {
      const int f    = c * 4096 + tid * 16;  // byte offset within 16 KB tile
      const int row  = f >> 7;               // 128 B per tile row
      const int col  = (f & 127) >> 2;       // float column
      const float* ga = A + (size_t)(m0 + row) * K + k0 + col;
      const float* gb = W + (size_t)(n0 + row) * K + k0 + col;
      async_copy_b128(ga, (const float*)((const char*)&As[s][0] + f));
      async_copy_b128(gb, (const float*)((const char*)&Bs[s][0] + f));
    }
  };

  const int NK = K / BK;
  issue_stage(0, 0);

  for (int kt = 0; kt < NK; ++kt) {
    const int s = kt & 1;
    if (kt + 1 < NK) {
      issue_stage((kt + 1) * BK, s ^ 1);  // start next stage's DMA first
      wait_async_le8();                   // oldest 8 (current stage) done
    } else {
      wait_async_le0();
    }
    __syncthreads();  // all threads' DMA for stage s visible

    v16bf af[2], bfv[4];
#pragma unroll
    for (int i = 0; i < 2; ++i) af[i] = ldsFragA_f32(&As[s][0], wm + i * 16);
#pragma unroll
    for (int j = 0; j < 4; ++j) bfv[j] = ldsFragB_f32(&Bs[s][0], wn + j * 16);
#pragma unroll
    for (int i = 0; i < 2; ++i)
#pragma unroll
      for (int j = 0; j < 4; ++j) acc[i][j].v = wmma_bf16(af[i], bfv[j], acc[i][j].v);

    __syncthreads();  // stage s fully consumed before it is refilled
  }

  // epilogue: C/D layout -> row = r + 8*(lane>=16), col = lane&15
  const int rb  = (lane & 16) ? 8 : 0;
  const int cof = lane & 15;
#pragma unroll
  for (int i = 0; i < 2; ++i) {
#pragma unroll
    for (int j = 0; j < 4; ++j) {
      const int gr0 = m0 + wm + i * 16 + rb;
      const int gc  = n0 + wn + j * 16 + cof;
      const float bval = bias[gc];
#pragma unroll
      for (int r = 0; r < 8; ++r) {
        float val = acc[i][j].f[r] + bval;
        if (RELU) val = fmaxf(val, 0.0f);
        C[(size_t)(gr0 + r) * N + gc] = val;
      }
    }
  }
}

// ---- Attention kernel -------------------------------------------------------
// One block (4 waves / 128 threads) per (b, h, 16-query-row tile).
// q,k,v in [B*S, D] layout; head h = columns [h*64, h*64+64).
__global__ __launch_bounds__(128) void attention_kernel(
    const float* __restrict__ q, const float* __restrict__ kk,
    const float* __restrict__ vv, const float* __restrict__ gammas,
    const int* __restrict__ maskp, float* __restrict__ outc) {
  __shared__ float sc[16 * SQ];  // 64 KB: full 16 x 1024 score block

  const int itile = blockIdx.x, h = blockIdx.y, b = blockIdx.z;
  const int i0   = itile * 16;
  const int tid  = threadIdx.x;
  const int lane = tid & 31;
  const int wid  = tid >> 5;
  const int maskv = maskp[0];
  const float gamma = -log1pf(__expf(gammas[h]));  // -softplus

  // --- scores = (q/8) @ k^T via WMMA; q fragments kept in registers ----------
  const size_t qbase =
      (size_t)(b * SQ + i0 + (lane & 15)) * DQ + h * DKQ + ((lane & 16) ? 16 : 0);
  const v16bf aq0 = globalFragA(q + qbase, 0.125f);        // K 0..31
  const v16bf aq1 = globalFragA(q + qbase + 32, 0.125f);   // K 32..63

  for (int t = 0; t < 16; ++t) {        // each wave: 16 of the 64 key tiles
    const int j0 = (wid * 16 + t) * 16;
    const size_t kbase =
        (size_t)(b * SQ + j0 + (lane & 15)) * DQ + h * DKQ + ((lane & 16) ? 16 : 0);
    const v16bf bk0 = globalFragB(kk + kbase);
    const v16bf bk1 = globalFragB(kk + kbase + 32);
    v8f c = {};
    c = wmma_bf16(aq0, bk0, c);
    c = wmma_bf16(aq1, bk1, c);
    AccU u; u.v = c;
    const int rb  = (lane & 16) ? 8 : 0;
    const int col = j0 + (lane & 15);
#pragma unroll
    for (int r = 0; r < 8; ++r) sc[(rb + r) * SQ + col] = u.f[r];
  }
  __syncthreads();

  // --- softmax + distance-decay rescale + final softmax ----------------------
  // 8 lanes per row; lane l owns contiguous block [l*128, l*128+128).
  const int row = tid >> 3;
  const int l   = tid & 7;
  const int gi  = i0 + row;
  float* srow   = sc + row * SQ;
  const int jb  = l * 128;

  float mx = -3.4e38f;
  for (int j = jb; j < jb + 128; ++j) mx = fmaxf(mx, srow[j]);
  for (int m = 1; m < 8; m <<= 1) mx = fmaxf(mx, __shfl_xor(mx, m, 32));

  float sum = 0.0f;
  for (int j = jb; j < jb + 128; ++j) sum += __expf(srow[j] - mx);
  for (int m = 1; m < 8; m <<= 1) sum += __shfl_xor(sum, m, 32);
  const float inv = 1.0f / sum;

  // masked-prob block sum, then exclusive suffix across the 8 lanes
  float bs = 0.0f;
  for (int j = jb; j < jb + 128; ++j) {
    const float p = __expf(srow[j] - mx) * inv;
    bs += ((j - gi) < maskv) ? p : 0.0f;
  }
  const int gb = lane & ~7;
  float suf = 0.0f;
  for (int m = 0; m < 8; ++m) {               // uniform control flow
    const float vbs = __shfl(bs, gb + m, 32);
    if (m > l) suf += vbs;
  }
  // rewrite scores with decay (iterate high->low so acc = suffix after j)
  float accs = suf;
  for (int j = jb + 127; j >= jb; --j) {
    const float s0 = srow[j];
    const float p  = __expf(s0 - mx) * inv;
    const bool allowed = (j - gi) < maskv;
    const float after = accs;                 // disttot - distcum[j]
    const float pos   = fabsf((float)(j - gi));
    const float dist  = sqrtf(fmaxf(after * pos, 0.0f));
    float eff = __expf(dist * gamma);
    eff = fminf(fmaxf(eff, 1e-5f), 1e5f);
    srow[j] = allowed ? s0 * eff : -1e20f;
    accs += allowed ? p : 0.0f;
  }

  // final softmax (masked entries exp-underflow to 0), optional zero-pad row 0
  float mx2 = -3.4e38f;
  for (int j = jb; j < jb + 128; ++j) mx2 = fmaxf(mx2, srow[j]);
  for (int m = 1; m < 8; m <<= 1) mx2 = fmaxf(mx2, __shfl_xor(mx2, m, 32));
  float sum2 = 0.0f;
  for (int j = jb; j < jb + 128; ++j) sum2 += __expf(srow[j] - mx2);
  for (int m = 1; m < 8; m <<= 1) sum2 += __shfl_xor(sum2, m, 32);
  const float inv2 = 1.0f / sum2;
  const float zfac = ((maskv == 0) && (gi == 0)) ? 0.0f : 1.0f;
  for (int j = jb; j < jb + 128; ++j)
    srow[j] = __expf(srow[j] - mx2) * inv2 * zfac;
  __syncthreads();

  // --- out = attn @ v via WMMA; each wave accumulates 8 of 32 key blocks -----
  AccU oacc[4];
#pragma unroll
  for (int n = 0; n < 4; ++n)
#pragma unroll
    for (int r = 0; r < 8; ++r) oacc[n].f[r] = 0.0f;

  const int rA  = lane & 15;
  const int kbA = (lane & 16) ? 16 : 0;
  const int kbB = (lane & 16) ? 16 : 0;
  for (int blk = wid * 8; blk < wid * 8 + 8; ++blk) {
    const int j0 = blk * 32;
    FragU fa;
    const float* sp = sc + rA * SQ + j0 + kbA;
#pragma unroll
    for (int e = 0; e < 8; ++e) {
      fa.h[e]     = (bf16_t)sp[e];
      fa.h[e + 8] = (bf16_t)sp[e + 8];
    }
#pragma unroll
    for (int n = 0; n < 4; ++n) {
      const float* vp =
          vv + (size_t)(b * SQ + j0 + kbB) * DQ + h * DKQ + n * 16 + (lane & 15);
      FragU fb;
#pragma unroll
      for (int e = 0; e < 16; ++e) fb.h[e] = (bf16_t)vp[(size_t)e * DQ];
      oacc[n].v = wmma_bf16(fa.v, fb.v, oacc[n].v);
    }
  }
  __syncthreads();  // probs no longer needed; reuse sc for partials

  {
    const int rb = (lane & 16) ? 8 : 0;
#pragma unroll
    for (int n = 0; n < 4; ++n)
#pragma unroll
      for (int r = 0; r < 8; ++r)
        sc[wid * 1024 + (rb + r) * 64 + n * 16 + (lane & 15)] = oacc[n].f[r];
  }
  __syncthreads();

  for (int idx = tid * 8; idx < tid * 8 + 8; ++idx) {
    const float s = sc[idx] + sc[1024 + idx] + sc[2048 + idx] + sc[3072 + idx];
    const int r = idx >> 6, c2 = idx & 63;
    outc[(size_t)(b * SQ + i0 + r) * DQ + h * DKQ + c2] = s;
  }
}

// ---- fused residual-add + LayerNorm (1 wave per row of 1024) ---------------
__global__ __launch_bounds__(128) void add_layernorm_kernel(
    const float* __restrict__ x, const float* __restrict__ y,
    const float* __restrict__ w, const float* __restrict__ bb,
    float* __restrict__ out, int N) {
  const int row  = blockIdx.x * 4 + (threadIdx.x >> 5);
  const int lane = threadIdx.x & 31;
  const float* xr = x + (size_t)row * N;
  const float* yr = y + (size_t)row * N;

  float s = 0.0f;
  for (int j = lane; j < N; j += 32) s += xr[j] + yr[j];
  for (int m = 16; m; m >>= 1) s += __shfl_xor(s, m, 32);
  const float mu = s / (float)N;

  float var = 0.0f;
  for (int j = lane; j < N; j += 32) {
    const float d = xr[j] + yr[j] - mu;
    var += d * d;
  }
  for (int m = 16; m; m >>= 1) var += __shfl_xor(var, m, 32);
  const float rstd = rsqrtf(var / (float)N + 1e-5f);

  float* orow = out + (size_t)row * N;
  for (int j = lane; j < N; j += 32)
    orow[j] = (xr[j] + yr[j] - mu) * rstd * w[j] + bb[j];
}

// ---- host orchestration -----------------------------------------------------
extern "C" void kernel_launch(void* const* d_in, const int* in_sizes, int n_in,
                              void* d_out, int out_size, void* d_ws, size_t ws_size,
                              hipStream_t stream) {
  (void)in_sizes; (void)n_in; (void)out_size; (void)ws_size;

  const int*   maskp  = (const int*)d_in[0];
  const float* query  = (const float*)d_in[1];
  const float* key    = (const float*)d_in[2];
  const float* values = (const float*)d_in[3];
  const float* Wq = (const float*)d_in[4];  const float* bq = (const float*)d_in[5];
  const float* Wk = (const float*)d_in[6];  const float* bk = (const float*)d_in[7];
  const float* Wv = (const float*)d_in[8];  const float* bv = (const float*)d_in[9];
  const float* Wo = (const float*)d_in[10]; const float* bo = (const float*)d_in[11];
  const float* gammas = (const float*)d_in[12];
  const float* ln1w = (const float*)d_in[13]; const float* ln1b = (const float*)d_in[14];
  const float* W1 = (const float*)d_in[15]; const float* b1 = (const float*)d_in[16];
  const float* W2 = (const float*)d_in[17]; const float* b2 = (const float*)d_in[18];
  const float* ln2w = (const float*)d_in[19]; const float* ln2b = (const float*)d_in[20];
  float* out = (float*)d_out;

  const int M = BQ * SQ;  // 4096 rows
  float* ws     = (float*)d_ws;
  float* qb     = ws;                          // 16 MB
  float* kb2    = qb  + (size_t)M * DQ;        // 16 MB
  float* vb     = kb2 + (size_t)M * DQ;        // 16 MB
  float* concat = vb  + (size_t)M * DQ;        // 16 MB
  float* x1     = concat + (size_t)M * DQ;     // 16 MB
  float* hidden = x1  + (size_t)M * DQ;        // 64 MB
  float* proj   = qb;    // reuse: q dead after attention
  float* ffnout = kb2;   // reuse: k dead after attention

  const dim3 blk(256);
  const dim3 gD(DQ / 128, M / 128);     // N=1024 GEMMs
  const dim3 gF(DFFQ / 128, M / 128);   // N=4096 GEMM

  gemm_bias_kernel<0><<<gD, blk, 0, stream>>>(query,  Wq, bq, qb,  M, DQ, DQ);
  gemm_bias_kernel<0><<<gD, blk, 0, stream>>>(key,    Wk, bk, kb2, M, DQ, DQ);
  gemm_bias_kernel<0><<<gD, blk, 0, stream>>>(values, Wv, bv, vb,  M, DQ, DQ);

  const dim3 ga(SQ / 16, HQ, BQ);
  attention_kernel<<<ga, 128, 0, stream>>>(qb, kb2, vb, gammas, maskp, concat);

  gemm_bias_kernel<0><<<gD, blk, 0, stream>>>(concat, Wo, bo, proj, M, DQ, DQ);
  add_layernorm_kernel<<<M / 4, 128, 0, stream>>>(query, proj, ln1w, ln1b, x1, DQ);

  gemm_bias_kernel<1><<<gF, blk, 0, stream>>>(x1, W1, b1, hidden, M, DFFQ, DQ);
  gemm_bias_kernel<0><<<gD, blk, 0, stream>>>(hidden, W2, b2, ffnout, M, DQ, DFFQ);
  add_layernorm_kernel<<<M / 4, 128, 0, stream>>>(x1, ffnout, ln2w, ln2b, out, DQ);
}
